// LocalOTLoss_60120952209498
// MI455X (gfx1250) — matrix operand order repaired and checked by reading, not compiled, not run on hardware
//
#include <hip/hip_runtime.h>

typedef __attribute__((ext_vector_type(16))) __bf16 v16bf;
typedef __attribute__((ext_vector_type(8)))  float  v8f;

#define EPS_F      0.1f
#define INV_EPS    10.0f
#define GAMMA_F    0.1f
#define GK         (GAMMA_F * INV_EPS)     /* 1.0f */
#define NEGK       (-1000000.0f * INV_EPS) /* -1e7 */
#define TAU_F      0.005f
#define DELTA_F    1e-9f

static constexpr int    BD   = 64;
static constexpr int    NV   = 512;
static constexpr int    NT   = 512;
static constexpr int    DIM  = 768;
static constexpr int    N1   = 513; // augmented

// ---- workspace layout (bytes) ----
static constexpr size_t OFF_VN   = 0;                                        // bf16 normalized v
static constexpr size_t OFF_TN   = OFF_VN  + (size_t)BD*NV*DIM*2;            // bf16 normalized t
static constexpr size_t OFF_A    = OFF_TN  + (size_t)BD*NT*DIM*2;            // f32 A (raw similarities)
static constexpr size_t OFF_U    = OFF_A   + (size_t)BD*NV*NT*4;             // u  [B][513]
static constexpr size_t OFF_V    = OFF_U   + (size_t)BD*N1*4;                // vv [B][513]
static constexpr size_t OFF_LMU  = OFF_V   + (size_t)BD*N1*4;                // log_mu
static constexpr size_t OFF_LNU  = OFF_LMU + (size_t)BD*N1*4;                // log_nu
static constexpr size_t OFF_RL   = OFF_LNU + (size_t)BD*N1*4;                // per-row loss
static constexpr size_t OFF_WVN  = OFF_RL  + (size_t)BD*N1*4;                // w_v numerator
static constexpr size_t OFF_WTN  = OFF_WVN + (size_t)BD*NV*4;                // w_t numerator
static constexpr size_t OFF_LOSS = OFF_WTN + (size_t)BD*NT*4;                // per-batch loss

__device__ __forceinline__ unsigned short f2bf(float f) {
    unsigned int u = __float_as_uint(f);
    unsigned int r = u + 0x7FFFu + ((u >> 16) & 1u); // RNE
    return (unsigned short)(r >> 16);
}

__device__ __forceinline__ float wave_sum(float v) {
#pragma unroll
    for (int off = 16; off; off >>= 1) v += __shfl_xor(v, off, 32);
    return v;
}

__device__ __forceinline__ float block_sum256(float v, float* sred) {
    __syncthreads();
    v = wave_sum(v);
    const int wave = threadIdx.x >> 5, lane = threadIdx.x & 31;
    if (lane == 0) sred[wave] = v;
    __syncthreads();
    if (wave == 0) {
        float w = (lane < 8) ? sred[lane] : 0.0f;
#pragma unroll
        for (int off = 4; off; off >>= 1) w += __shfl_xor(w, off, 32);
        if (lane == 0) sred[0] = w;
    }
    __syncthreads();
    return sred[0];
}

// ---------- 1) normalize rows and convert to bf16 ----------
__global__ __launch_bounds__(256) void ot_normalize(const float* __restrict__ v,
                                                    const float* __restrict__ t,
                                                    unsigned short* __restrict__ vn,
                                                    unsigned short* __restrict__ tn) {
    const int row = blockIdx.x, b = blockIdx.y, which = blockIdx.z;
    const float* src = (which ? t : v) + ((size_t)b * 512 + row) * DIM;
    unsigned short* dst = (which ? tn : vn) + ((size_t)b * 512 + row) * DIM;
    __shared__ float sred[8];
    float x0 = src[threadIdx.x];
    float x1 = src[threadIdx.x + 256];
    float x2 = src[threadIdx.x + 512];
    float ss = block_sum256(x0 * x0 + x1 * x1 + x2 * x2, sred);
    float inv = 1.0f / fmaxf(sqrtf(ss), 1e-12f);
    dst[threadIdx.x]       = f2bf(x0 * inv);
    dst[threadIdx.x + 256] = f2bf(x1 * inv);
    dst[threadIdx.x + 512] = f2bf(x2 * inv);
}

// ---------- 2) masks -> log_mu/log_nu ; init u,vv ----------
__global__ __launch_bounds__(256) void ot_masklog(const unsigned char* __restrict__ vmask,
                                                  const unsigned char* __restrict__ tmask,
                                                  float* __restrict__ logmu, float* __restrict__ lognu,
                                                  float* __restrict__ u, float* __restrict__ vv) {
    const int b = blockIdx.x;
    __shared__ float sred[8];
    float cv = 0.f, ct = 0.f;
    for (int i = threadIdx.x; i < 512; i += 256) {
        cv += (vmask[b * 512 + i] ? 1.f : 0.f);
        ct += (tmask[b * 512 + i] ? 1.f : 0.f);
    }
    float cntv = block_sum256(cv, sred);
    float cntt = block_sum256(ct, sred);
    for (int i = threadIdx.x; i < N1; i += 256) {
        float lm, ln;
        if (i == 512) { lm = logf(1.0f + 1e-9f); ln = lm; }
        else {
            float mv = vmask[b * 512 + i] ? (1.0f / (cntv + 1e-9f)) : 0.0f;
            float mt = tmask[b * 512 + i] ? (1.0f / (cntt + 1e-9f)) : 0.0f;
            lm = logf(mv + 1e-9f);
            ln = logf(mt + 1e-9f);
        }
        logmu[b * N1 + i] = lm;
        lognu[b * N1 + i] = ln;
        u[b * N1 + i] = 0.0f;
        vv[b * N1 + i] = 0.0f;
    }
}

// ---------- 3) bf16 WMMA GEMM: A[b] = vn[b] (512x768) * tn[b]^T ----------
__global__ __launch_bounds__(256) void ot_gemm(const unsigned short* __restrict__ vn,
                                               const unsigned short* __restrict__ tn,
                                               float* __restrict__ A) {
    const int b    = blockIdx.z;
    const int m_wg = blockIdx.x * 64;
    const int n_wg = blockIdx.y * 256;
    const int wave = threadIdx.x >> 5;
    const int lane = threadIdx.x & 31;
    const int m_w  = m_wg + (wave & 1) * 32;
    const int n_w  = n_wg + (wave >> 1) * 64;
    const unsigned short* Vb = vn + (size_t)b * 512 * DIM;
    const unsigned short* Tb = tn + (size_t)b * 512 * DIM;
    const int am = lane & 15;
    const int ak = (lane < 16) ? 0 : 8;   // A: K chunks {ak..ak+8, ak+16..ak+24}
    const int bn = lane & 15;
    const int bk = (lane < 16) ? 0 : 8;   // B: K chunks {bk..bk+8, bk+16..bk+24}

    union ABfrag { v16bf v; uint4 q[2]; };
    const v8f zero = {0.f, 0.f, 0.f, 0.f, 0.f, 0.f, 0.f, 0.f};
    v8f c[2][4];
#pragma unroll
    for (int i = 0; i < 2; i++)
#pragma unroll
        for (int j = 0; j < 4; j++) c[i][j] = zero;

    for (int k0 = 0; k0 < DIM; k0 += 32) {
        ABfrag a[2];
#pragma unroll
        for (int i = 0; i < 2; i++) {
            const unsigned short* p = Vb + (size_t)(m_w + i * 16 + am) * DIM + k0 + ak;
            a[i].q[0] = *(const uint4*)(p);
            a[i].q[1] = *(const uint4*)(p + 16);
        }
        ABfrag bf[4];
#pragma unroll
        for (int j = 0; j < 4; j++) {
            const unsigned short* p = Tb + (size_t)(n_w + j * 16 + bn) * DIM + k0 + bk;
            bf[j].q[0] = *(const uint4*)(p);
            bf[j].q[1] = *(const uint4*)(p + 16);
        }
#pragma unroll
        for (int i = 0; i < 2; i++)
#pragma unroll
            for (int j = 0; j < 4; j++)
                c[i][j] = __builtin_amdgcn_wmma_f32_16x16x32_bf16(
                    false, a[i].v, false, bf[j].v, (short)0, c[i][j], false, false);
    }

    float* Ab = A + (size_t)b * 512 * 512;
#pragma unroll
    for (int i = 0; i < 2; i++) {
        const int rbase = m_w + i * 16 + ((lane < 16) ? 0 : 8);
#pragma unroll
        for (int r = 0; r < 8; r++) {
            const int row = rbase + r;
#pragma unroll
            for (int j = 0; j < 4; j++)
                Ab[(size_t)row * 512 + n_w + j * 16 + (lane & 15)] = c[i][j][r];
        }
    }
}

// ---------- 4) Sinkhorn row pass: u = log_mu - LSE_j(K + vv) ----------
__global__ __launch_bounds__(256) void ot_row_lse(const float* __restrict__ A,
                                                  const unsigned char* __restrict__ vmask,
                                                  const unsigned char* __restrict__ tmask,
                                                  const float* __restrict__ logmu,
                                                  const float* __restrict__ vv,
                                                  float* __restrict__ u) {
    const int wave = threadIdx.x >> 5, lane = threadIdx.x & 31;
    const int i = blockIdx.x * 8 + wave;
    if (i > 512) return;
    const int b = blockIdx.y;
    const bool rowAug = (i == 512);
    const bool vm = rowAug ? true : (vmask[b * 512 + i] != 0);
    const float* Arow = A + ((size_t)b * 512 + (rowAug ? 0 : i)) * 512;
    const float* vvb = vv + b * N1;
    float m = -1e30f, s = 0.f;
    for (int j = lane; j < N1; j += 32) {
        float x;
        if (j == 512 || rowAug) x = GK + vvb[j];
        else {
            float k = (vm && tmask[b * 512 + j]) ? Arow[j] * INV_EPS : NEGK;
            x = k + vvb[j];
        }
        float M = fmaxf(m, x);
        s = s * __expf(m - M) + __expf(x - M);
        m = M;
    }
#pragma unroll
    for (int off = 16; off; off >>= 1) {
        float mo = __shfl_xor(m, off, 32);
        float so = __shfl_xor(s, off, 32);
        float M = fmaxf(m, mo);
        s = s * __expf(m - M) + so * __expf(mo - M);
        m = M;
    }
    if (lane == 0) u[b * N1 + i] = logmu[b * N1 + i] - (m + logf(s));
}

// ---------- 5) Sinkhorn col pass: vv = log_nu - LSE_i(K + u) ----------
__global__ __launch_bounds__(256) void ot_col_lse(const float* __restrict__ A,
                                                  const unsigned char* __restrict__ vmask,
                                                  const unsigned char* __restrict__ tmask,
                                                  const float* __restrict__ lognu,
                                                  const float* __restrict__ u,
                                                  float* __restrict__ vv) {
    const int j = blockIdx.x * 256 + threadIdx.x;
    if (j > 512) return;
    const int b = blockIdx.y;
    const bool colAug = (j == 512);
    const bool tm = colAug ? true : (tmask[b * 512 + j] != 0);
    const float* ub = u + b * N1;
    const float* Ab = A + (size_t)b * 512 * 512;
    float m = -1e30f, s = 0.f;
    for (int i = 0; i < N1; i++) {
        float x;
        if (i == 512 || colAug) x = GK + ub[i];
        else {
            float k = ((vmask[b * 512 + i] != 0) && tm) ? Ab[(size_t)i * 512 + j] * INV_EPS : NEGK;
            x = k + ub[i];
        }
        float M = fmaxf(m, x);
        s = s * __expf(m - M) + __expf(x - M);
        m = M;
    }
    vv[b * N1 + j] = lognu[b * N1 + j] - (m + logf(s));
}

// ---------- 6) finalize, row-oriented: per-row loss + w_v numerators ----------
__global__ __launch_bounds__(256) void ot_fin_row(const float* __restrict__ A,
                                                  const unsigned char* __restrict__ vmask,
                                                  const unsigned char* __restrict__ tmask,
                                                  const float* __restrict__ u,
                                                  const float* __restrict__ vv,
                                                  float* __restrict__ rowloss,
                                                  float* __restrict__ wvn) {
    const int wave = threadIdx.x >> 5, lane = threadIdx.x & 31;
    const int i = blockIdx.x * 8 + wave;
    if (i > 512) return;
    const int b = blockIdx.y;
    const bool rowAug = (i == 512);
    const bool vm = rowAug ? true : (vmask[b * 512 + i] != 0);
    const float* Arow = A + ((size_t)b * 512 + (rowAug ? 0 : i)) * 512;
    const float ui = u[b * N1 + i];
    const float* vvb = vv + b * N1;
    float accL = 0.f, accW = 0.f;
    for (int j = lane; j < N1; j += 32) {
        float k, raw;
        if (j == 512 || rowAug) { k = GK; raw = GAMMA_F; }
        else {
            raw = Arow[j];
            k = (vm && tmask[b * 512 + j]) ? raw * INV_EPS : NEGK;
        }
        float T = __expf(ui + vvb[j] + k);
        accL += T * (1.0f - raw);
        if (!rowAug && j < 512) accW += fmaxf(T - TAU_F, 0.0f);
    }
    accL = wave_sum(accL);
    accW = wave_sum(accW);
    if (lane == 0) {
        rowloss[b * N1 + i] = accL;
        if (!rowAug) wvn[b * 512 + i] = accW + DELTA_F;
    }
}

// ---------- 7) finalize, col-oriented: w_t numerators ----------
__global__ __launch_bounds__(256) void ot_fin_col(const float* __restrict__ A,
                                                  const unsigned char* __restrict__ vmask,
                                                  const unsigned char* __restrict__ tmask,
                                                  const float* __restrict__ u,
                                                  const float* __restrict__ vv,
                                                  float* __restrict__ wtn) {
    const int j = blockIdx.x * 256 + threadIdx.x;
    if (j >= 512) return;
    const int b = blockIdx.y;
    const bool tm = (tmask[b * 512 + j] != 0);
    const float* Ab = A + (size_t)b * 512 * 512;
    const float* ub = u + b * N1;
    const float vj = vv[b * N1 + j];
    float acc = 0.f;
    for (int i = 0; i < 512; i++) {
        float k = ((vmask[b * 512 + i] != 0) && tm) ? Ab[(size_t)i * 512 + j] * INV_EPS : NEGK;
        float T = __expf(ub[i] + vj + k);
        acc += fmaxf(T - TAU_F, 0.0f);
    }
    wtn[b * 512 + j] = acc + DELTA_F;
}

// ---------- 8) per-batch: loss reduce + normalize w_v, w_t ----------
__global__ __launch_bounds__(256) void ot_fin_batch(const float* __restrict__ rowloss,
                                                    const float* __restrict__ wvn,
                                                    const float* __restrict__ wtn,
                                                    float* __restrict__ lossb,
                                                    float* __restrict__ out_wv,
                                                    float* __restrict__ out_wt) {
    const int b = blockIdx.x;
    __shared__ float sred[8];
    float l = 0.f;
    for (int i = threadIdx.x; i < N1; i += 256) l += rowloss[b * N1 + i];
    l = block_sum256(l, sred);
    if (threadIdx.x == 0) lossb[b] = l;
    float sv = 0.f, st = 0.f;
    for (int i = threadIdx.x; i < 512; i += 256) {
        sv += wvn[b * 512 + i];
        st += wtn[b * 512 + i];
    }
    sv = block_sum256(sv, sred);
    st = block_sum256(st, sred);
    for (int i = threadIdx.x; i < 512; i += 256) {
        out_wv[b * 512 + i] = wvn[b * 512 + i] / sv;
        out_wt[b * 512 + i] = wtn[b * 512 + i] / st;
    }
}

// ---------- 9) mean loss ----------
__global__ void ot_fin_mean(const float* __restrict__ lossb, float* __restrict__ out0) {
    if (threadIdx.x == 0) {
        float s = 0.f;
        for (int b = 0; b < BD; b++) s += lossb[b];
        out0[0] = s / (float)BD;
    }
}

extern "C" void kernel_launch(void* const* d_in, const int* in_sizes, int n_in,
                              void* d_out, int out_size, void* d_ws, size_t ws_size,
                              hipStream_t stream) {
    const float*         v     = (const float*)d_in[0];
    const float*         t     = (const float*)d_in[1];
    const unsigned char* vmask = (const unsigned char*)d_in[2];
    const unsigned char* tmask = (const unsigned char*)d_in[3];
    float* out = (float*)d_out;

    char* ws = (char*)d_ws;
    unsigned short* vn    = (unsigned short*)(ws + OFF_VN);
    unsigned short* tn    = (unsigned short*)(ws + OFF_TN);
    float*          A     = (float*)(ws + OFF_A);
    float*          u     = (float*)(ws + OFF_U);
    float*          vv    = (float*)(ws + OFF_V);
    float*          logmu = (float*)(ws + OFF_LMU);
    float*          lognu = (float*)(ws + OFF_LNU);
    float*          rl    = (float*)(ws + OFF_RL);
    float*          wvn   = (float*)(ws + OFF_WVN);
    float*          wtn   = (float*)(ws + OFF_WTN);
    float*          lossb = (float*)(ws + OFF_LOSS);

    ot_normalize<<<dim3(512, BD, 2), 256, 0, stream>>>(v, t, vn, tn);
    ot_masklog<<<dim3(BD), 256, 0, stream>>>(vmask, tmask, logmu, lognu, u, vv);
    ot_gemm<<<dim3(8, 2, BD), 256, 0, stream>>>(vn, tn, A);
    for (int it = 0; it < 5; it++) {
        ot_row_lse<<<dim3(65, BD), 256, 0, stream>>>(A, vmask, tmask, logmu, vv, u);
        ot_col_lse<<<dim3(3, BD), 256, 0, stream>>>(A, vmask, tmask, lognu, u, vv);
    }
    ot_fin_row<<<dim3(65, BD), 256, 0, stream>>>(A, vmask, tmask, u, vv, rl, wvn);
    ot_fin_col<<<dim3(2, BD), 256, 0, stream>>>(A, vmask, tmask, u, vv, wtn);
    ot_fin_batch<<<dim3(BD), 256, 0, stream>>>(rl, wvn, wtn, lossb, out + 1, out + 1 + BD * NV);
    ot_fin_mean<<<1, 64, 0, stream>>>(lossb, out);
}
